// PitchDiffusion_62689342653113
// MI455X (gfx1250) — matrix operand order repaired and checked by reading, not compiled, not run on hardware
//
#include <hip/hip_runtime.h>
#include <hip/hip_bf16.h>

// ---------------- problem constants ----------------
#define BB   8
#define TT   2048
#define INC  384
#define HH   256
#define TD   128
#define NLAY 4
#define NST  100
#define MM   (BB*TT)      // 16384 flattened rows

typedef __attribute__((ext_vector_type(16))) __bf16 v16bf;
typedef __attribute__((ext_vector_type(8)))  float  v8f;

union FragAB { v16bf v; uint4 q[2]; };

__device__ __forceinline__ unsigned short f2bf(float f) {
    union { float f; unsigned u; } v; v.f = f;
    unsigned r = v.u + 0x7FFFu + ((v.u >> 16) & 1u);
    return (unsigned short)(r >> 16);
}
__device__ __forceinline__ float bf2f(unsigned short h) {
    union { unsigned u; float f; } v; v.u = ((unsigned)h) << 16; return v.f;
}
__device__ __forceinline__ float gelu_erf(float x) {
    return 0.5f * x * (1.0f + erff(x * 0.70710678118654752440f));
}

#define WMMA_BF16(A, B, C) __builtin_amdgcn_wmma_f32_16x16x32_bf16( \
    false, (A), false, (B), (short)0, (C), false, false)

// ---------------- prep kernels ----------------

__global__ void sched_kernel(float* sched) {
    if (threadIdx.x == 0 && blockIdx.x == 0) {
        float ab = 1.0f;
        for (int i = 0; i < NST; ++i) {
            float beta = 1e-4f + (0.02f - 1e-4f) * (float)i / (float)(NST - 1);
            float alpha = 1.0f - beta;
            ab *= alpha;
            sched[i*3+0] = beta; sched[i*3+1] = alpha; sched[i*3+2] = ab;
        }
    }
}

__global__ void cond_kernel(const float* __restrict__ enc, const float* __restrict__ mask,
                            unsigned short* __restrict__ out) {
    int idx = blockIdx.x * 256 + threadIdx.x;
    if (idx >= MM * INC) return;
    int row = idx / INC;
    out[idx] = f2bf(enc[idx] * mask[row]);
}

// dst[n*K + k] = bf16(src[k*N + n])
__global__ void transpose_kernel(const float* __restrict__ src, unsigned short* __restrict__ dst,
                                 int K, int N) {
    int idx = blockIdx.x * 256 + threadIdx.x;
    if (idx >= K * N) return;
    int k = idx / N, n = idx % N;
    dst[(size_t)n * K + k] = f2bf(src[idx]);
}

// ---------------- generic WMMA GEMM: Out = act(A @ B^T + bias) ----------------
// 32 M-rows per block (two 16-row subtiles per wave) -> each B fragment feeds 2 WMMAs.
// grid = M/32 blocks x 128 threads (4 waves); wave w -> cols [64w, 64w+64)
template<int KIN, bool GELU>
__global__ void gemm_bias_act(const unsigned short* __restrict__ A,
                              const unsigned short* __restrict__ BT,
                              const float* __restrict__ bias,
                              unsigned short* __restrict__ Out) {
    const int lane = threadIdx.x & 31;
    const int wv   = threadIdx.x >> 5;
    const int hi   = lane >> 4;
    const int ml   = lane & 15;
    const int row0 = blockIdx.x * 32;

    v8f acc0[4], acc1[4];
    #pragma unroll
    for (int f = 0; f < 4; ++f) { acc0[f] = (v8f){0,0,0,0,0,0,0,0}; acc1[f] = acc0[f]; }

    for (int c = 0; c < KIN / 32; ++c) {
        FragAB a0, a1;
        size_t ab0 = (size_t)(row0 + ml) * KIN + c * 32 + hi * 8;
        size_t ab1 = (size_t)(row0 + 16 + ml) * KIN + c * 32 + hi * 8;
        a0.q[0] = *(const uint4*)&A[ab0];
        a0.q[1] = *(const uint4*)&A[ab0 + 16];
        a1.q[0] = *(const uint4*)&A[ab1];
        a1.q[1] = *(const uint4*)&A[ab1 + 16];
        #pragma unroll
        for (int f = 0; f < 4; ++f) {
            FragAB b;
            int n = wv * 64 + f * 16 + ml;
            size_t wo = (size_t)n * KIN + c * 32 + hi * 8;
            b.q[0] = *(const uint4*)&BT[wo];
            b.q[1] = *(const uint4*)&BT[wo + 16];
            acc0[f] = WMMA_BF16(a0.v, b.v, acc0[f]);
            acc1[f] = WMMA_BF16(a1.v, b.v, acc1[f]);
        }
    }

    #pragma unroll
    for (int f = 0; f < 4; ++f) {
        int n = wv * 64 + f * 16 + ml;
        float bn = bias[n];
        #pragma unroll
        for (int r = 0; r < 8; ++r) {
            int mr = r + hi * 8;
            float v0 = acc0[f][r] + bn;
            float v1 = acc1[f][r] + bn;
            if (GELU) { v0 = gelu_erf(v0); v1 = gelu_erf(v1); }
            Out[(size_t)(row0 + mr) * HH + n]      = f2bf(v0);
            Out[(size_t)(row0 + 16 + mr) * HH + n] = f2bf(v1);
        }
    }
}

// ---------------- conv(k=3) + bias + LayerNorm (+optional GELU) via WMMA ----------------
// 32 output rows per block, staged with 2-row halo in LDS (rows t0-1 .. t0+32).
// HAS_SPLIT: input row = [ Xh(256) | Xco(256)+tvec ] (KIN=512), else Xh only (KIN=256).
// WT: [3][256 out][KIN] bf16.  Out: [M][256] bf16.
template<int KIN, bool HAS_SPLIT, bool GELU_OUT>
__global__ void conv_ln_kernel(const unsigned short* __restrict__ Xh,
                               const unsigned short* __restrict__ Xco,
                               const float* __restrict__ tvec,
                               const unsigned short* __restrict__ WT,
                               const float* __restrict__ bias,
                               const float* __restrict__ g,
                               const float* __restrict__ be,
                               unsigned short* __restrict__ Out) {
    constexpr int LDSTRIDE = KIN + 8;                 // bank-conflict padding
    __shared__ unsigned short xi[34 * LDSTRIDE];      // rows t0-1 .. t0+32
    __shared__ float rsum[32], rsum2[32];

    const int tile = blockIdx.x;
    const int tloc = (tile & (TT / 32 - 1)) * 32;     // t within batch
    const int row0 = tile * 32;                       // flattened global row

    // ---- stage A tile (halo, zero pad at batch edges, fold tvec into co half) ----
    const int nChunk = KIN / 8;                       // uint4 chunks per row
    for (int ch = threadIdx.x; ch < 34 * nChunk; ch += blockDim.x) {
        int lr = ch / nChunk;
        int colq = (ch % nChunk) * 8;
        int t = tloc + lr - 1;
        uint4 val;
        if (t < 0 || t >= TT) {
            val = (uint4){0u, 0u, 0u, 0u};
        } else {
            int grow = row0 + lr - 1;
            if (!HAS_SPLIT || colq < HH) {
                val = *(const uint4*)&Xh[(size_t)grow * HH + colq];
            } else {
                int cc = colq - HH;
                uint4 c4 = *(const uint4*)&Xco[(size_t)grow * HH + cc];
                unsigned short* cu = (unsigned short*)&c4;
                #pragma unroll
                for (int j = 0; j < 8; ++j)
                    cu[j] = f2bf(bf2f(cu[j]) + tvec[cc + j]);
                val = c4;
            }
        }
        *(uint4*)&xi[lr * LDSTRIDE + colq] = val;
    }
    if (threadIdx.x < 32) { rsum[threadIdx.x] = 0.0f; rsum2[threadIdx.x] = 0.0f; }
    __syncthreads();

    const int lane = threadIdx.x & 31;
    const int wv   = threadIdx.x >> 5;
    const int hi   = lane >> 4;
    const int ml   = lane & 15;

    v8f acc0[4], acc1[4];
    #pragma unroll
    for (int f = 0; f < 4; ++f) { acc0[f] = (v8f){0,0,0,0,0,0,0,0}; acc1[f] = acc0[f]; }

    // ---- K loop: 3 taps x KIN/32 chunks; each B fragment feeds both M-subtiles ----
    #pragma unroll
    for (int k = 0; k < 3; ++k) {
        for (int c = 0; c < KIN / 32; ++c) {
            FragAB a0, a1;
            int ab0 = (ml + k) * LDSTRIDE + c * 32 + hi * 8;
            int ab1 = (16 + ml + k) * LDSTRIDE + c * 32 + hi * 8;
            a0.q[0] = *(const uint4*)&xi[ab0];
            a0.q[1] = *(const uint4*)&xi[ab0 + 16];
            a1.q[0] = *(const uint4*)&xi[ab1];
            a1.q[1] = *(const uint4*)&xi[ab1 + 16];
            #pragma unroll
            for (int f = 0; f < 4; ++f) {
                FragAB b;
                int n = wv * 64 + f * 16 + ml;
                size_t wo = (size_t)(k * HH + n) * KIN + c * 32 + hi * 8;
                b.q[0] = *(const uint4*)&WT[wo];
                b.q[1] = *(const uint4*)&WT[wo + 16];
                if (f == 0 && c + 2 < KIN / 32)
                    __builtin_prefetch(&WT[wo + 64], 0, 0);   // global_prefetch_b8
                acc0[f] = WMMA_BF16(a0.v, b.v, acc0[f]);
                acc1[f] = WMMA_BF16(a1.v, b.v, acc1[f]);
            }
        }
    }

    // ---- bias + per-row mean/var over all 256 channels ----
    float p0[8], p20[8], p1[8], p21[8];
    #pragma unroll
    for (int r = 0; r < 8; ++r) { p0[r] = p20[r] = p1[r] = p21[r] = 0.0f; }
    #pragma unroll
    for (int f = 0; f < 4; ++f) {
        float bn = bias[wv * 64 + f * 16 + ml];
        #pragma unroll
        for (int r = 0; r < 8; ++r) {
            float v0 = acc0[f][r] + bn;
            float v1 = acc1[f][r] + bn;
            acc0[f][r] = v0; acc1[f][r] = v1;
            p0[r] += v0; p20[r] += v0 * v0;
            p1[r] += v1; p21[r] += v1 * v1;
        }
    }
    #pragma unroll
    for (int m = 1; m < 16; m <<= 1) {
        #pragma unroll
        for (int r = 0; r < 8; ++r) {
            p0[r]  += __shfl_xor(p0[r],  m, 32);
            p20[r] += __shfl_xor(p20[r], m, 32);
            p1[r]  += __shfl_xor(p1[r],  m, 32);
            p21[r] += __shfl_xor(p21[r], m, 32);
        }
    }
    if (ml == 0) {   // lanes 0 (rows r+0) and 16 (rows r+8) of each subtile
        #pragma unroll
        for (int r = 0; r < 8; ++r) {
            atomicAdd(&rsum[r + hi * 8],       p0[r]);
            atomicAdd(&rsum2[r + hi * 8],      p20[r]);
            atomicAdd(&rsum[16 + r + hi * 8],  p1[r]);
            atomicAdd(&rsum2[16 + r + hi * 8], p21[r]);
        }
    }
    __syncthreads();

    // ---- normalize, scale/shift, optional GELU, store bf16 ----
    #pragma unroll
    for (int f = 0; f < 4; ++f) {
        int n = wv * 64 + f * 16 + ml;
        float gn = g[n], bn2 = be[n];
        #pragma unroll
        for (int r = 0; r < 8; ++r) {
            int mr = r + hi * 8;
            float mean0 = rsum[mr] * (1.0f / HH);
            float var0  = rsum2[mr] * (1.0f / HH) - mean0 * mean0;
            float v0 = (acc0[f][r] - mean0) * rsqrtf(var0 + 1e-5f) * gn + bn2;
            float mean1 = rsum[16 + mr] * (1.0f / HH);
            float var1  = rsum2[16 + mr] * (1.0f / HH) - mean1 * mean1;
            float v1 = (acc1[f][r] - mean1) * rsqrtf(var1 + 1e-5f) * gn + bn2;
            if (GELU_OUT) { v0 = gelu_erf(v0); v1 = gelu_erf(v1); }
            Out[(size_t)(row0 + mr) * HH + n]      = f2bf(v0);
            Out[(size_t)(row0 + 16 + mr) * HH + n] = f2bf(v1);
        }
    }
}

// ---------------- per-step small kernels ----------------

__global__ void temb_kernel(const float* __restrict__ LWt, const float* __restrict__ Lbt,
                            float t, float* __restrict__ tvec) {
    __shared__ float te[TD];
    if (threadIdx.x < TD / 2) {
        float fr = expf(-logf(10000.0f) / (float)(TD / 2 - 1) * (float)threadIdx.x);
        float a = t * fr;
        te[threadIdx.x]          = sinf(a);
        te[threadIdx.x + TD / 2] = cosf(a);
    }
    __syncthreads();
    int c = threadIdx.x;
    #pragma unroll
    for (int l = 0; l < NLAY; ++l) {
        float s = Lbt[l * HH + c];
        for (int d = 0; d < TD; ++d)
            s += te[d] * LWt[((size_t)l * TD + d) * HH + c];
        tvec[l * HH + c] = gelu_erf(s);
    }
}

__global__ void hinit_kernel(const float* __restrict__ x, const float* __restrict__ Wx,
                             const float* __restrict__ bx, unsigned short* __restrict__ H) {
    int row = blockIdx.x, c = threadIdx.x;
    H[(size_t)row * HH + c] = f2bf(x[row] * Wx[c] + bx[c]);
}

__global__ void update_kernel(const unsigned short* __restrict__ Hb,
                              const float* __restrict__ Wo, const float* __restrict__ bo,
                              float* __restrict__ x, const float* __restrict__ noise,
                              const float* __restrict__ sched, int t) {
    int row = blockIdx.x * blockDim.x + threadIdx.x;
    if (row >= MM) return;
    float s = bo[0];
    const uint4* hp = (const uint4*)&Hb[(size_t)row * HH];
    for (int q = 0; q < HH / 8; ++q) {
        uint4 v = hp[q];
        const unsigned short* cu = (const unsigned short*)&v;
        #pragma unroll
        for (int j = 0; j < 8; ++j) s += bf2f(cu[j]) * Wo[q * 8 + j];
    }
    float beta = sched[t*3], alpha = sched[t*3+1], ab = sched[t*3+2];
    float xn = (x[row] - beta * rsqrtf(1.0f - ab) * s) * rsqrtf(alpha);
    if (t > 0) xn += sqrtf(beta) * noise[row];
    x[row] = xn;
}

__global__ void final_kernel(const float* __restrict__ x, const float* __restrict__ mask,
                             float* __restrict__ out) {
    int row = blockIdx.x * blockDim.x + threadIdx.x;
    if (row < MM) out[row] = x[row] * mask[row];
}

// ---------------- host ----------------
extern "C" void kernel_launch(void* const* d_in, const int* in_sizes, int n_in,
                              void* d_out, int out_size, void* d_ws, size_t ws_size,
                              hipStream_t stream) {
    const float* enc   = (const float*)d_in[0];
    const float* mask  = (const float*)d_in[1];
    const float* x_init= (const float*)d_in[2];
    const float* noise = (const float*)d_in[3];
    const float* Wc0   = (const float*)d_in[4];
    const float* bc0   = (const float*)d_in[5];
    const float* Wx    = (const float*)d_in[6];
    const float* bx    = (const float*)d_in[7];
    const float* Wo    = (const float*)d_in[8];
    const float* bo    = (const float*)d_in[9];
    const float* LWc   = (const float*)d_in[10];
    const float* Lbc   = (const float*)d_in[11];
    const float* LWt   = (const float*)d_in[12];
    const float* Lbt   = (const float*)d_in[13];
    const float* LK1   = (const float*)d_in[14];
    const float* Lb1   = (const float*)d_in[15];
    const float* LK2   = (const float*)d_in[16];
    const float* Lb2   = (const float*)d_in[17];
    const float* Lg1   = (const float*)d_in[18];
    const float* Lbe1  = (const float*)d_in[19];
    const float* Lg2   = (const float*)d_in[20];
    const float* Lbe2  = (const float*)d_in[21];

    char* wp = (char*)d_ws;
    auto alloc = [&](size_t bytes) { char* p = wp; wp += (bytes + 255) & ~(size_t)255; return p; };
    float*          sched  = (float*)alloc((size_t)NST * 3 * 4);
    float*          xbuf   = (float*)alloc((size_t)MM * 4);
    float*          tvec   = (float*)alloc((size_t)NLAY * HH * 4);
    unsigned short* condbf = (unsigned short*)alloc((size_t)MM * INC * 2);
    unsigned short* condh  = (unsigned short*)alloc((size_t)MM * HH * 2);
    unsigned short* cobase = (unsigned short*)alloc((size_t)NLAY * MM * HH * 2);
    unsigned short* Hbuf   = (unsigned short*)alloc((size_t)MM * HH * 2);
    unsigned short* Ybuf   = (unsigned short*)alloc((size_t)MM * HH * 2);
    unsigned short* W1T    = (unsigned short*)alloc((size_t)NLAY * 3 * HH * (2 * HH) * 2);
    unsigned short* W2T    = (unsigned short*)alloc((size_t)NLAY * 3 * HH * HH * 2);
    unsigned short* WcT    = (unsigned short*)alloc((size_t)NLAY * HH * HH * 2);
    unsigned short* Wc0T   = (unsigned short*)alloc((size_t)HH * INC * 2);

    // ---- prep ----
    sched_kernel<<<1, 1, 0, stream>>>(sched);
    cond_kernel<<<(MM * INC + 255) / 256, 256, 0, stream>>>(enc, mask, condbf);
    for (int l = 0; l < NLAY; ++l)
        for (int k = 0; k < 3; ++k) {
            transpose_kernel<<<(2*HH*HH + 255) / 256, 256, 0, stream>>>(
                LK1 + ((size_t)(l*3 + k)) * (2*HH) * HH, W1T + ((size_t)(l*3 + k)) * HH * (2*HH),
                2*HH, HH);
            transpose_kernel<<<(HH*HH + 255) / 256, 256, 0, stream>>>(
                LK2 + ((size_t)(l*3 + k)) * HH * HH, W2T + ((size_t)(l*3 + k)) * HH * HH,
                HH, HH);
        }
    for (int l = 0; l < NLAY; ++l)
        transpose_kernel<<<(HH*HH + 255) / 256, 256, 0, stream>>>(
            LWc + (size_t)l * HH * HH, WcT + (size_t)l * HH * HH, HH, HH);
    transpose_kernel<<<(INC*HH + 255) / 256, 256, 0, stream>>>(Wc0, Wc0T, INC, HH);

    // ---- step-invariant GEMMs (WMMA): cond_h, then co_base[l] = gelu(cond_h@Wc_l + bc_l) ----
    gemm_bias_act<INC, false><<<MM / 32, 128, 0, stream>>>(condbf, Wc0T, bc0, condh);
    for (int l = 0; l < NLAY; ++l)
        gemm_bias_act<HH, true><<<MM / 32, 128, 0, stream>>>(
            condh, WcT + (size_t)l * HH * HH, Lbc + (size_t)l * HH,
            cobase + (size_t)l * MM * HH);

    hipMemcpyAsync(xbuf, x_init, (size_t)MM * 4, hipMemcpyDeviceToDevice, stream);

    // ---- 100 reverse diffusion steps ----
    for (int s = 0; s < NST; ++s) {
        int t = NST - 1 - s;
        temb_kernel<<<1, HH, 0, stream>>>(LWt, Lbt, (float)t, tvec);
        hinit_kernel<<<MM, HH, 0, stream>>>(xbuf, Wx, bx, Hbuf);
        for (int l = 0; l < NLAY; ++l) {
            conv_ln_kernel<2*HH, true, true><<<MM / 32, 128, 0, stream>>>(
                Hbuf, cobase + (size_t)l * MM * HH, tvec + (size_t)l * HH,
                W1T + (size_t)l * 3 * HH * (2*HH),
                Lb1 + (size_t)l * HH, Lg1 + (size_t)l * HH, Lbe1 + (size_t)l * HH, Ybuf);
            conv_ln_kernel<HH, false, false><<<MM / 32, 128, 0, stream>>>(
                Ybuf, nullptr, nullptr,
                W2T + (size_t)l * 3 * HH * HH,
                Lb2 + (size_t)l * HH, Lg2 + (size_t)l * HH, Lbe2 + (size_t)l * HH, Hbuf);
        }
        update_kernel<<<MM / 256, 256, 0, stream>>>(Hbuf, Wo, bo, xbuf,
                                                    noise + (size_t)t * MM, sched, t);
    }

    final_kernel<<<MM / 256, 256, 0, stream>>>(xbuf, mask, (float*)d_out);
}
